// TitansL2_60902636257296
// MI455X (gfx1250) — compile-verified, matrix operand order checked by reading, not run on hardware
//
#include <hip/hip_runtime.h>
#include <hip/hip_bf16.h>
#include <stdint.h>

// ---------------------------------------------------------------------------
// Titans-L2 delta-rule attention for MI455X (gfx1250, wave32, WMMA + TDM)
//   B=4, T=2048, C=1024, H=16, D=64
// GEMMs in bf16 via v_wmma_f32_16x16x32_bf16 (f32 accumulate); K panels are
// staged into LDS by the Tensor Data Mover (tensor_load_to_lds, TENSORcnt),
// double-buffered against the WMMA compute. Sequential scan kept in f32.
// ---------------------------------------------------------------------------

typedef __attribute__((ext_vector_type(16))) __bf16 v16bf;
typedef __attribute__((ext_vector_type(8)))  float  v8f;
typedef __attribute__((ext_vector_type(4)))  unsigned int v4u;
typedef __attribute__((ext_vector_type(8)))  int  v8i_;
typedef __attribute__((ext_vector_type(4)))  int  v4i_;

#define Bdim 4
#define Tdim 2048
#define Cdim 1024
#define Hdim 16
#define Ddim 64
#define Mrows (Bdim * Tdim)   // 8192

// workspace layout (units: uint16_t / bf16 elements)
#define WS_XBF   ((size_t)0)                       // x bf16:        8192*1024
#define WS_WQKV  ((size_t)8388608)                 // [Wq;Wk;Wv]:    3*1024*1024
#define WS_WPROJ ((size_t)11534336)                // Wproj bf16:    1024*1024
#define WS_QKV   ((size_t)12582912)                // q,k,v bf16:    3*8192*1024
#define WS_YBF   ((size_t)37748736)                // y bf16:        8192*1024
// total: 46,137,344 elems = ~88 MB

__device__ __forceinline__ uint16_t f2bf(float f) {
  uint32_t u = __builtin_bit_cast(uint32_t, f);
  uint32_t r = u + 0x7FFFu + ((u >> 16) & 1u);   // round-to-nearest-even
  return (uint16_t)(r >> 16);
}
__device__ __forceinline__ float bf2f(uint16_t h) {
  return __builtin_bit_cast(float, (uint32_t)h << 16);
}

// --------------------------- f32 -> bf16 convert ---------------------------
__global__ __launch_bounds__(256) void k_f32_to_bf16(
    const float* __restrict__ src, uint16_t* __restrict__ dst, int n) {
  int i = blockIdx.x * 256 + threadIdx.x;
  int stride = gridDim.x * 256;
  for (; i < n; i += stride) dst[i] = f2bf(src[i]);
}

// --------------------------- TDM panel loader ------------------------------
// DMA a 128-row x 32-col bf16 panel (row stride 1024 elems in memory) into
// LDS with 32B of padding after every 64B row => LDS row stride 96B (48 bf16),
// matching the WMMA fragment loader. Issued by ONE wave; tracked by TENSORcnt.
//
// D# group0: count=1, lds_addr, global_addr(57b), type=2.
// D# group1: data_size=2B, pad_enable, pad_interval=3 (16 DW = 64B),
//            pad_amount=7 (8 DW = 32B), tensor_dim0=1024, tensor_dim1=8192,
//            tile_dim0=32, tile_dim1=128, tensor_dim0_stride=1024.
__device__ __forceinline__ void tdm_load_panel(const uint16_t* gsrc,
                                               uint32_t lds_byte_addr) {
  uint64_t ga = (uint64_t)(uintptr_t)gsrc;
  v4u g0;
  g0[0] = 1u;                                                  // count=1
  g0[1] = lds_byte_addr;                                       // lds_addr
  g0[2] = (uint32_t)ga;                                        // global[31:0]
  g0[3] = (uint32_t)((ga >> 32) & 0x1FFFFFFu) | 0x80000000u;   // global[56:32] | type=2
  v8i_ g1;
  g1[0] = (1 << 16) | (1 << 20) | (3 << 22) | (7 << 25);       // dsz=2B, pad 64B->+32B
  g1[1] = (int)(((unsigned)1024 & 0xFFFFu) << 16);             // tensor_dim0 lo16
  g1[2] = (int)(((unsigned)8192 & 0xFFFFu) << 16);             // tensor_dim1 lo16
  g1[3] = (int)((unsigned)32 << 16);                           // tile_dim0 = 32
  g1[4] = 128;                                                 // tile_dim1 = 128
  g1[5] = 1024;                                                // dim0_stride = 1024
  g1[6] = 0;
  g1[7] = 0;
  v4i_ z4 = {0, 0, 0, 0};
#if __has_include(<hip/amd_detail/amd_gfx1250_TDM.h>)
  v8i_ z8 = {0, 0, 0, 0, 0, 0, 0, 0};
  __builtin_amdgcn_tensor_load_to_lds(g0, g1, z4, z4, z8, 0);  // 6-arg toolchain
#else
  __builtin_amdgcn_tensor_load_to_lds(g0, g1, z4, z4, 0);      // ROCm 7.2 (5-arg)
#endif
}

// ------------------------------- bf16 GEMM ---------------------------------
// C[m,n] = sum_k A[m,k] * W[n,k]   (TN: both operands K-contiguous)
// Block tile 128x128, 8 waves; wave tile 64x32 = 4x2 WMMA 16x16 tiles.
// K panels (128x32 bf16) staged into LDS by the TDM, double-buffered.
// MODE 0: fused qkv (N=3072), bf16 scatter epilogue. MODE 1: f32 epilogue.
template <int MODE>
__global__ __launch_bounds__(256) void k_gemm_bf16(
    const uint16_t* __restrict__ A, const uint16_t* __restrict__ W,
    uint16_t* __restrict__ outBf, float* __restrict__ outF) {
  constexpr int K    = 1024;
  constexpr int BK   = 32;
  constexpr int LDSS = 48;   // bf16 elems per LDS row (64B data + 32B pad)

  __shared__ uint16_t lAs[2][128 * LDSS];
  __shared__ uint16_t lBs[2][128 * LDSS];

  const int tid  = threadIdx.x;
  const int lane = tid & 31;
  const int wave = tid >> 5;
  const int wm   = wave >> 2;          // 0..1 -> 64-row slab
  const int wn   = wave & 3;           // 0..3 -> 32-col slab
  const int m0   = blockIdx.y * 128;
  const int n0   = blockIdx.x * 128;

  union Frag { v16bf v; uint4 q[2]; };

  v8f acc[4][2];
#pragma unroll
  for (int mi = 0; mi < 4; ++mi)
#pragma unroll
    for (int ni = 0; ni < 2; ++ni)
      acc[mi][ni] = (v8f){0.f, 0.f, 0.f, 0.f, 0.f, 0.f, 0.f, 0.f};

  // preload panel 0: wave0 -> A panel, wave1 -> B panel
  if (tid < 32) {
    tdm_load_panel(A + (size_t)m0 * K, (uint32_t)(uintptr_t)&lAs[0][0]);
  } else if (tid < 64) {
    tdm_load_panel(W + (size_t)n0 * K, (uint32_t)(uintptr_t)&lBs[0][0]);
  }

  int buf = 0;
  for (int k0 = 0; k0 < K; k0 += BK) {
    // issuing waves drain their DMA; barrier publishes the panel to all waves
    if (tid < 64) __builtin_amdgcn_s_wait_tensorcnt(0);
    __syncthreads();

    // kick off next panel into the other buffer while we compute this one
    if (k0 + BK < K) {
      if (tid < 32) {
        tdm_load_panel(A + (size_t)m0 * K + (k0 + BK),
                       (uint32_t)(uintptr_t)&lAs[buf ^ 1][0]);
      } else if (tid < 64) {
        tdm_load_panel(W + (size_t)n0 * K + (k0 + BK),
                       (uint32_t)(uintptr_t)&lBs[buf ^ 1][0]);
      }
    }

    // Fragment loads. 16-bit A 16x32 lane layout (ISA 7.12.2):
    //   lane<16 : K = {khalf..khalf+7, 16+khalf..}, khalf = (lane>>4)*8
    const int khalf = (lane >> 4) * 8;
    Frag af[4], bfr[2];
#pragma unroll
    for (int mi = 0; mi < 4; ++mi) {
      int row = wm * 64 + mi * 16 + (lane & 15);
      af[mi].q[0] = *(const uint4*)&lAs[buf][row * LDSS + khalf];
      af[mi].q[1] = *(const uint4*)&lAs[buf][row * LDSS + 16 + khalf];
    }
#pragma unroll
    for (int ni = 0; ni < 2; ++ni) {
      int coln = wn * 32 + ni * 16 + (lane & 15);
      bfr[ni].q[0] = *(const uint4*)&lBs[buf][coln * LDSS + khalf];
      bfr[ni].q[1] = *(const uint4*)&lBs[buf][coln * LDSS + 16 + khalf];
    }

#pragma unroll
    for (int mi = 0; mi < 4; ++mi)
#pragma unroll
      for (int ni = 0; ni < 2; ++ni)
        acc[mi][ni] = __builtin_amdgcn_wmma_f32_16x16x32_bf16(
            false, af[mi].v, false, bfr[ni].v, (short)0, acc[mi][ni],
            false, false);

    buf ^= 1;
  }

  // Epilogue. C layout: VGPR r -> row r + (lane>=16 ? 8 : 0), col lane&15.
#pragma unroll
  for (int mi = 0; mi < 4; ++mi) {
#pragma unroll
    for (int ni = 0; ni < 2; ++ni) {
#pragma unroll
      for (int r = 0; r < 8; ++r) {
        int gm = m0 + wm * 64 + mi * 16 + (lane >> 4) * 8 + r;
        int gn = n0 + wn * 32 + ni * 16 + (lane & 15);
        float val = acc[mi][ni][r];
        if constexpr (MODE == 0) {
          int p  = gn >> 10;       // 0=q, 1=k, 2=v
          int cc = gn & 1023;
          outBf[(size_t)p * Mrows * 1024 + (size_t)gm * 1024 + cc] = f2bf(val);
        } else {
          outF[(size_t)gm * 1024 + gn] = val;
        }
      }
    }
  }
}

// ------------------------------ delta-rule scan ----------------------------
// One block per (b,h); 64x64 f32 state resident in LDS.
// Per step t: normalize k_t; y = S q_t; Mk = S k_t;
//             S += (beta*v_t - alpha*Mk) k_t^T  (rank-1 update).
__global__ __launch_bounds__(256) void k_scan(
    const uint16_t* __restrict__ qkv, const float* __restrict__ alpha_raw,
    const float* __restrict__ beta_raw, const float* __restrict__ state0,
    uint16_t* __restrict__ ybf, float* __restrict__ stateOut) {
  __shared__ float S[64 * 64];
  __shared__ float qb[64], kb[64], vbuf[64], cb[64];
  __shared__ float red[2];

  const int tid = threadIdx.x;
  const int bh  = blockIdx.x;      // b*H + h
  const int b   = bh >> 4;
  const int h   = bh & 15;

  const float alpha = 0.5f / (1.f + __expf(-alpha_raw[h]));
  const float beta  = 0.5f / (1.f + __expf(-beta_raw[h]));

  const uint16_t* qg = qkv;
  const uint16_t* kg = qkv + (size_t)Mrows * 1024;
  const uint16_t* vg = qkv + (size_t)2 * Mrows * 1024;

  for (int i = tid; i < 4096; i += 256) S[i] = state0[(size_t)bh * 4096 + i];
  __syncthreads();

  const int d  = tid >> 2;          // 0..63: state row
  const int es = (tid & 3) * 16;    // 16-wide e-chunk within the row
  float* Srow = &S[d * 64];

  for (int t = 0; t < Tdim; ++t) {
    const size_t rowbase = ((size_t)(b * Tdim + t)) * 1024 + (size_t)h * 64;

    if (tid < 64) {                 // load q/k/v row; wave-reduce ||k||^2
      float qv_ = bf2f(qg[rowbase + tid]);
      float kv_ = bf2f(kg[rowbase + tid]);
      float vv_ = bf2f(vg[rowbase + tid]);
      qb[tid] = qv_; kb[tid] = kv_; vbuf[tid] = vv_;
      float p = kv_ * kv_;
#pragma unroll
      for (int off = 16; off >= 1; off >>= 1) p += __shfl_xor(p, off, 32);
      if ((tid & 31) == 0) red[tid >> 5] = p;
    }
    __syncthreads();
    if (tid < 64) {                 // F.normalize(k, eps=1e-12)
      float nrm = sqrtf(red[0] + red[1]);
      kb[tid] = kb[tid] / fmaxf(nrm, 1e-12f);
    }
    __syncthreads();

    float sy = 0.f, sm = 0.f;       // y[d], Mk[d] partials (4 lanes per d)
#pragma unroll
    for (int e = 0; e < 16; ++e) {
      float s = Srow[es + e];
      sy = fmaf(s, qb[es + e], sy);
      sm = fmaf(s, kb[es + e], sm);
    }
    sy += __shfl_xor(sy, 1, 32); sy += __shfl_xor(sy, 2, 32);
    sm += __shfl_xor(sm, 1, 32); sm += __shfl_xor(sm, 2, 32);
    if ((tid & 3) == 0) {
      cb[d] = beta * vbuf[d] - alpha * sm;
      ybf[rowbase + d] = f2bf(sy);  // y in (B,T,C) layout for out-projection
    }
    __syncthreads();

    const float cd = cb[d];         // rank-1 update
#pragma unroll
    for (int e = 0; e < 16; ++e)
      Srow[es + e] = fmaf(cd, kb[es + e], Srow[es + e]);
    __syncthreads();
  }

  for (int i = tid; i < 4096; i += 256)
    stateOut[(size_t)bh * 4096 + i] = S[i];
}

// ------------------------------- launcher ----------------------------------
extern "C" void kernel_launch(void* const* d_in, const int* in_sizes, int n_in,
                              void* d_out, int out_size, void* d_ws,
                              size_t ws_size, hipStream_t stream) {
  const float* x  = (const float*)d_in[0];
  const float* Wq = (const float*)d_in[1];
  const float* Wk = (const float*)d_in[2];
  const float* Wv = (const float*)d_in[3];
  const float* Wp = (const float*)d_in[4];
  const float* ar = (const float*)d_in[5];
  const float* br = (const float*)d_in[6];
  const float* st = (const float*)d_in[7];
  float* out = (float*)d_out;
  uint16_t* ws = (uint16_t*)d_ws;

  auto cvt = [&](const float* s, size_t off, int n) {
    int blocks = (n + 2047) / 2048;
    k_f32_to_bf16<<<blocks, 256, 0, stream>>>(s, ws + off, n);
  };
  cvt(x,  WS_XBF,              Mrows * 1024);
  cvt(Wq, WS_WQKV,             1024 * 1024);
  cvt(Wk, WS_WQKV + 1048576,   1024 * 1024);
  cvt(Wv, WS_WQKV + 2097152,   1024 * 1024);
  cvt(Wp, WS_WPROJ,            1024 * 1024);

  // fused q/k/v projection: (8192x1024) x (3072x1024)^T
  dim3 gq(3072 / 128, Mrows / 128);
  k_gemm_bf16<0><<<gq, 256, 0, stream>>>(ws + WS_XBF, ws + WS_WQKV,
                                         ws + WS_QKV, nullptr);

  // sequential delta-rule scan; writes y (bf16) + final state (f32 tail)
  k_scan<<<Bdim * Hdim, 256, 0, stream>>>(ws + WS_QKV, ar, br, st,
                                          ws + WS_YBF,
                                          out + (size_t)Mrows * 1024);

  // output projection: (8192x1024) x (1024x1024)^T -> f32 d_out
  dim3 gp(1024 / 128, Mrows / 128);
  k_gemm_bf16<1><<<gp, 256, 0, stream>>>(ws + WS_YBF, ws + WS_WPROJ, nullptr,
                                         out);
}
